// LSTM_1589137899980
// MI455X (gfx1250) — matrix operand order repaired and checked by reading, not compiled, not run on hardware
//
#include <hip/hip_runtime.h>
#include <hip/hip_bf16.h>

// ---------------------------------------------------------------------------
// LSTM (B=128, T=1024, D=256, H=512) + dense head on gfx1250 (MI455X).
// Persistent-scan design: c in VGPRs, weights in LDS (staged via TDM
// tensor_load_to_lds), recurrent operand staged via async-to-LDS loads,
// bf16 WMMA GEMMs, grid barrier between timesteps, double-buffered bf16 h.
// ---------------------------------------------------------------------------

typedef __attribute__((ext_vector_type(16))) __bf16 v16bf;
typedef __attribute__((ext_vector_type(8)))  float  v8f;
typedef __attribute__((ext_vector_type(4)))  unsigned int v4u;
typedef __attribute__((ext_vector_type(8)))  int    v8i_;
typedef __attribute__((ext_vector_type(4)))  int    v4i_;

#define B_    128
#define T_    1024
#define D_    256
#define H_    512
#define G_    2048      // 4*H
#define O_    256
#define KC_   768       // H + D  (concatenated K: [h | x])
#define KCH_  128       // K chunk staged in LDS
#define NCH_  6         // KC_ / KCH_
#define NWG_  32        // persistent workgroups; each owns 16 cols of H
#define COLS_ 16

// workspace layout (bytes)
#define WS_BAR   0
#define WS_H0    256
#define WS_H1    (WS_H0 + B_*H_*2)
#define WS_WCAT  (WS_H1 + B_*H_*2)                    // bf16 [G_][KC_] col-major (per gate-col)
#define WS_WDCM  (WS_WCAT + (size_t)G_*KC_*2)         // bf16 [O_][H_]  col-major

union Frag { uint4 u[2]; v16bf v; };

__device__ __forceinline__ unsigned short f2bf(float f) {
  unsigned u = __float_as_uint(f);
  unsigned r = u + 0x7FFFu + ((u >> 16) & 1u);   // round-to-nearest-even
  return (unsigned short)(r >> 16);
}
__device__ __forceinline__ float sigmf(float x)    { return 1.0f / (1.0f + __expf(-x)); }
__device__ __forceinline__ float tanhfast(float x) { return 2.0f / (1.0f + __expf(-2.0f * x)) - 1.0f; }

// Sense-reversing device-wide barrier (NWG_ resident WGs; trivially resident
// on MI455X). bar[0]=arrival count, bar[1]=generation. Reset by prep kernel.
__device__ __forceinline__ void grid_barrier(int* bar) {
  __threadfence();
  __syncthreads();
  if (threadIdx.x == 0) {
    volatile int* gen = bar + 1;
    int g = *gen;
    if (atomicAdd(bar, 1) == NWG_ - 1) {
      atomicExch(bar, 0);
      __threadfence();
      atomicAdd((int*)gen, 1);
    } else {
      while (*gen == g) { __builtin_amdgcn_s_sleep(2); }
    }
  }
  __syncthreads();
}

// ---------------------------------------------------------------------------
// Kernel 0: weight prep. Build col-major bf16 concatenated weights
//   wcat[n][k] : k<H -> Wh[k][n], k>=H -> Wi[k-H][n]     (n over G_ gate-cols)
//   wdcm[n][k] = Wd[k][n]                                (head weights)
// and reset the grid-barrier words (ws is not re-poisoned between replays).
// ---------------------------------------------------------------------------
__global__ void lstm_prep(const float* __restrict__ Wi, const float* __restrict__ Wh,
                          const float* __restrict__ Wd,
                          unsigned short* __restrict__ wcat,
                          unsigned short* __restrict__ wdcm,
                          int* __restrict__ bar) {
  const size_t stride = (size_t)gridDim.x * blockDim.x;
  const size_t idx = (size_t)blockIdx.x * blockDim.x + threadIdx.x;
  if (idx == 0) { bar[0] = 0; bar[1] = 0; }
  const size_t n1 = (size_t)G_ * KC_;
  for (size_t i = idx; i < n1; i += stride) {
    const size_t n = i / KC_, k = i % KC_;
    const float v = (k < H_) ? Wh[k * G_ + n] : Wi[(k - H_) * G_ + n];
    wcat[n * KC_ + k] = f2bf(v);
  }
  const size_t n2 = (size_t)O_ * H_;
  for (size_t i = idx; i < n2; i += stride) {
    const size_t n = i / H_, k = i % H_;
    wdcm[n * H_ + k] = f2bf(Wd[k * O_ + n]);
  }
}

// ---------------------------------------------------------------------------
// Kernel 1: persistent recurrent scan.
// Grid = 32 WGs x 256 threads (8 waves). WG wg owns H-columns
// [wg*16, wg*16+16) across all 4 gates. Wave w owns batch-row tile
// [w*16, w*16+16). Per step: Y = [h|x] @ Wcat  (K=768) via bf16 WMMA,
// gate nonlinearities, c update in registers, bf16 h_new -> global,
// grid barrier, next step.
// LDS: Wlds bf16[64][768] (96KB, staged once via TDM) + Alds bf16[128][128].
// ---------------------------------------------------------------------------
__global__ void __launch_bounds__(256, 1) lstm_scan(
    const float* __restrict__ X, const float* __restrict__ bias,
    unsigned short* __restrict__ h0buf, unsigned short* __restrict__ h1buf,
    const unsigned short* __restrict__ wcat, int* __restrict__ bar) {
  extern __shared__ unsigned short smem[];
  unsigned short* Wlds = smem;              // [64][KC_]
  unsigned short* Alds = smem + 64 * KC_;   // [B_][KCH_]

  const int tid  = threadIdx.x;
  const int wave = tid >> 5;
  const int lane = tid & 31;
  const int half = lane >> 4;
  const int ln   = lane & 15;
  const int colBase = blockIdx.x * COLS_;

  // ---- stage this WG's weight slice into LDS (once) -----------------------
  // Four contiguous 24KB blocks (16 gate-cols x 768 K, col-major) -> four 1-D
  // TDM descriptors issued by wave 0; tracked with TENSORcnt.
#if __has_builtin(__builtin_amdgcn_tensor_load_to_lds)
  if (wave == 0) {
    for (int gate = 0; gate < 4; ++gate) {
      const unsigned elems = 16u * KC_;   // 12288 bf16 = 24576 bytes
      const unsigned lds   = (unsigned)(uintptr_t)(Wlds + gate * 16 * KC_);
      const unsigned long long ga =
          (unsigned long long)(uintptr_t)(wcat + (size_t)(gate * H_ + colBase) * KC_);
      // D# group0: count=1 | lds_addr | 57-bit global_addr | type=2
      const v4u g0 = { 1u,
                       lds,
                       (unsigned)(ga & 0xFFFFFFFFu),
                       (unsigned)((ga >> 32) & 0x1FFFFFFu) | (2u << 30) };
      // D# group1: data_size=2B; tensor_dim0=tile_dim0=elems; tensor_dim1=1;
      //            tensor_dim0_stride=elems; everything else 0.
      const v8i_ g1 = { (int)(1u << 16),
                        (int)((elems & 0xFFFFu) << 16),
                        (int)(((elems >> 16) & 0xFFFFu) | (1u << 16)),
                        (int)((elems & 0xFFFFu) << 16),
                        0,
                        (int)elems,
                        0, 0 };
      const v4i_ gz = { 0, 0, 0, 0 };
#if __has_include(<hip/amd_detail/amd_gfx1250_TDM.h>)
      const v8i_ gz8 = { 0, 0, 0, 0, 0, 0, 0, 0 };
      __builtin_amdgcn_tensor_load_to_lds(g0, g1, gz, gz, gz8, 0);
#else
      __builtin_amdgcn_tensor_load_to_lds(g0, g1, gz, gz, 0);
#endif
    }
    __builtin_amdgcn_s_wait_tensorcnt(0);
  }
#else
  {
    const int c = tid >> 2;            // local col 0..63  (gate*16 + j)
    const int q = tid & 3;             // quarter of the K-column
    const int gate = c >> 4, j = c & 15;
    const int gcol = gate * H_ + colBase + j;
    const uint4* src = (const uint4*)(wcat + (size_t)gcol * KC_ + q * 192);
    uint4* dst = (uint4*)(Wlds + c * KC_ + q * 192);
#pragma unroll
    for (int i = 0; i < 24; ++i) dst[i] = src[i];
  }
#endif

  // ---- zero our slice of h0 (t=0 reads h0) --------------------------------
  if (tid < B_) {
    const uint4 z = {0u, 0u, 0u, 0u};
    uint4* d = (uint4*)(h0buf + (size_t)tid * H_ + colBase);
    d[0] = z; d[1] = z;
  }

  // per-lane bias (gate order i,f,g,o over columns of G_)
  const int col = colBase + ln;
  const float bi = bias[0 * H_ + col];
  const float bf = bias[1 * H_ + col];
  const float bg = bias[2 * H_ + col];
  const float bo = bias[3 * H_ + col];

  float creg[8];
#pragma unroll
  for (int e = 0; e < 8; ++e) creg[e] = 0.f;

  grid_barrier(bar);

  for (int t = 0; t < T_; ++t) {
    const unsigned short* hr = (t & 1) ? h1buf : h0buf;  // read  h_{t}
    unsigned short*       hw = (t & 1) ? h0buf : h1buf;  // write h_{t+1}

    v8f acc0 = {}; v8f acc1 = {}; v8f acc2 = {}; v8f acc3 = {};

    for (int ch = 0; ch < NCH_; ++ch) {
      __syncthreads();   // protect Alds from previous chunk's readers
      // ---- stage A chunk [128 rows x 128 K] bf16 --------------------------
      {
        const int r = tid >> 1, hh = tid & 1;
        if (ch < 4) {
          // h part: K in [0,512). Pure bf16 copy -> async global->LDS loads
          // (ASYNCcnt path; one INST_OFFSET advances both LDS and global addr).
          const unsigned ldsoff =
              (unsigned)(uintptr_t)(Alds + r * KCH_ + hh * 64);
          const unsigned long long ga =
              (unsigned long long)(uintptr_t)(hr + (size_t)r * H_ + ch * KCH_ + hh * 64);
          asm volatile(
              "global_load_async_to_lds_b128 %0, %1, off\n\t"
              "global_load_async_to_lds_b128 %0, %1, off offset:16\n\t"
              "global_load_async_to_lds_b128 %0, %1, off offset:32\n\t"
              "global_load_async_to_lds_b128 %0, %1, off offset:48\n\t"
              "global_load_async_to_lds_b128 %0, %1, off offset:64\n\t"
              "global_load_async_to_lds_b128 %0, %1, off offset:80\n\t"
              "global_load_async_to_lds_b128 %0, %1, off offset:96\n\t"
              "global_load_async_to_lds_b128 %0, %1, off offset:112"
              :: "v"(ldsoff), "v"(ga)
              : "memory");
          asm volatile("s_wait_asynccnt 0" ::: "memory");
        } else {
          // x part: K in [512,768) -> X[:, t, :] fp32 -> bf16 (needs VALU cvt)
          uint2* dst = (uint2*)(Alds + r * KCH_ + hh * 64);
          const float* xp = X + ((size_t)r * T_ + t) * D_ + (ch - 4) * KCH_ + hh * 64;
          const float4* src = (const float4*)xp;
#pragma unroll
          for (int i = 0; i < 16; ++i) {
            const float4 v = src[i];
            uint2 p;
            p.x = (unsigned)f2bf(v.x) | ((unsigned)f2bf(v.y) << 16);
            p.y = (unsigned)f2bf(v.z) | ((unsigned)f2bf(v.w) << 16);
            dst[i] = p;
          }
          if (t + 1 < T_) __builtin_prefetch(xp + D_, 0, 0);  // global_prefetch_b8
        }
      }
      __syncthreads();

      // ---- WMMA over this chunk: 4 k-steps x 4 gate tiles -----------------
      const int m = wave * 16 + ln;   // batch row held by this lane (A layout)
#pragma unroll
      for (int k32 = 0; k32 < 4; ++k32) {
        Frag a;
        a.u[0] = *(const uint4*)(Alds + m * KCH_ + k32 * 32 + 16 * half);
        a.u[1] = *(const uint4*)(Alds + m * KCH_ + k32 * 32 + 16 * half + 8);
        const unsigned short* wb = Wlds + k32 * 32 + 16 * half + ln * KC_;
        Frag w0, w1, w2, w3;
        { const uint4* q = (const uint4*)(wb + 0  * 16 * KC_); w0.u[0] = q[0]; w0.u[1] = q[1]; }
        { const uint4* q = (const uint4*)(wb + 1  * 16 * KC_); w1.u[0] = q[0]; w1.u[1] = q[1]; }
        { const uint4* q = (const uint4*)(wb + 2  * 16 * KC_); w2.u[0] = q[0]; w2.u[1] = q[1]; }
        { const uint4* q = (const uint4*)(wb + 3  * 16 * KC_); w3.u[0] = q[0]; w3.u[1] = q[1]; }
        acc0 = __builtin_amdgcn_wmma_f32_16x16x32_bf16(false, a.v, false, w0.v, (short)0, acc0, false, false);
        acc1 = __builtin_amdgcn_wmma_f32_16x16x32_bf16(false, a.v, false, w1.v, (short)0, acc1, false, false);
        acc2 = __builtin_amdgcn_wmma_f32_16x16x32_bf16(false, a.v, false, w2.v, (short)0, acc2, false, false);
        acc3 = __builtin_amdgcn_wmma_f32_16x16x32_bf16(false, a.v, false, w3.v, (short)0, acc3, false, false);
      }
    }

    // ---- gates + cell update (all four gate values live in this lane) -----
#pragma unroll
    for (int e = 0; e < 8; ++e) {
      const float yi = acc0[e] + bi;
      const float yf = acc1[e] + bf;
      const float yg = acc2[e] + bg;
      const float yo = acc3[e] + bo;
      const float ig = sigmf(yi);
      const float fg = sigmf(yf);
      const float gg = tanhfast(yg);
      const float og = sigmf(yo);
      const float c  = fg * creg[e] + ig * gg;
      creg[e] = c;
      const float h  = og * tanhfast(c);
      // C/D layout: element e -> M = e + 8*half (within tile), N = ln
      hw[(size_t)(wave * 16 + e + 8 * half) * H_ + col] = f2bf(h);
    }

    grid_barrier(bar);
  }
  // final h_T ends in h0buf (T_ even)
}

// ---------------------------------------------------------------------------
// Kernel 2: head — out = h_T @ Wd + bd.  One 16x16 tile per wave, K=512.
// 128 tiles -> 16 blocks x 8 waves.
// ---------------------------------------------------------------------------
__global__ void __launch_bounds__(256, 4) lstm_head(
    const unsigned short* __restrict__ hT, const unsigned short* __restrict__ wdcm,
    const float* __restrict__ bd, float* __restrict__ out) {
  const int tid = threadIdx.x, wave = tid >> 5, lane = tid & 31;
  const int half = lane >> 4, ln = lane & 15;
  const int tile = blockIdx.x * 8 + wave;      // 0..127
  const int rt = tile >> 4, ct = tile & 15;    // 8 row tiles x 16 col tiles
  v8f acc = {};
  const int m = rt * 16 + ln;
#pragma unroll 4
  for (int k32 = 0; k32 < 16; ++k32) {
    Frag a, w;
    a.u[0] = *(const uint4*)(hT + (size_t)m * H_ + k32 * 32 + 16 * half);
    a.u[1] = *(const uint4*)(hT + (size_t)m * H_ + k32 * 32 + 16 * half + 8);
    const uint4* q = (const uint4*)(wdcm + (size_t)(ct * 16 + ln) * H_ + k32 * 32 + 16 * half);
    w.u[0] = q[0]; w.u[1] = q[1];
    acc = __builtin_amdgcn_wmma_f32_16x16x32_bf16(false, a.v, false, w.v, (short)0, acc, false, false);
  }
  const int n = ct * 16 + ln;
  const float bv = bd[n];
#pragma unroll
  for (int e = 0; e < 8; ++e)
    out[(size_t)(rt * 16 + e + 8 * half) * O_ + n] = acc[e] + bv;
}

// ---------------------------------------------------------------------------
extern "C" void kernel_launch(void* const* d_in, const int* in_sizes, int n_in,
                              void* d_out, int out_size, void* d_ws, size_t ws_size,
                              hipStream_t stream) {
  (void)in_sizes; (void)n_in; (void)out_size; (void)ws_size;
  const float* X  = (const float*)d_in[0];
  const float* Wi = (const float*)d_in[1];
  const float* Wh = (const float*)d_in[2];
  const float* b  = (const float*)d_in[3];
  const float* Wd = (const float*)d_in[4];
  const float* bd = (const float*)d_in[5];

  char* ws = (char*)d_ws;
  int* bar             = (int*)(ws + WS_BAR);
  unsigned short* h0   = (unsigned short*)(ws + WS_H0);
  unsigned short* h1   = (unsigned short*)(ws + WS_H1);
  unsigned short* wcat = (unsigned short*)(ws + WS_WCAT);
  unsigned short* wdcm = (unsigned short*)(ws + WS_WDCM);

  lstm_prep<<<1024, 256, 0, stream>>>(Wi, Wh, Wd, wcat, wdcm, bar);
  const size_t lds_bytes = (size_t)(64 * KC_ + B_ * KCH_) * sizeof(unsigned short); // 128 KB
  lstm_scan<<<NWG_, 256, lds_bytes, stream>>>(X, b, h0, h1, wcat, bar);
  lstm_head<<<16, 256, 0, stream>>>(h0, wdcm, bd, (float*)d_out);
}
